// MoEFFN_4191888081459
// MI455X (gfx1250) — matrix operand order repaired and checked by reading, not compiled, not run on hardware
//
#include <hip/hip_runtime.h>
#include <hip/hip_bf16.h>

// ---------------- problem constants ----------------
static constexpr int D      = 2048;   // d_model
static constexpr int HID    = 5632;   // hidden
static constexpr int NEXP   = 8;
static constexpr int TOKENS = 4096;   // 4*1024
static constexpr int BM = 64, BN = 128, BK = 32;
static constexpr int NTHREADS = 256;  // 8 waves
static constexpr int LDA = 40;        // A LDS k-stride (bf16), 80B rows -> 16B aligned
static constexpr int LDB = 40;        // B LDS k-stride (bf16), 80B rows -> 16B aligned
static constexpr int CAP_ROWS  = 8704;           // 8192 assignments + 8*64 padding
static constexpr int MAX_TILES = CAP_ROWS / BM;  // 136 (fixed worst-case grid)

// ---------------- workspace layout (bytes) ----------------
static constexpr size_t WS_COUNTS = 0;                       // 8  int
static constexpr size_t WS_CURSOR = 64;                      // 8  int
static constexpr size_t WS_TOT    = 128;                     // 1  int
static constexpr size_t WS_TILE_E = 192;                     // 136 int
static constexpr size_t WS_TOK_E  = 1280;                    // 4096*2 int
static constexpr size_t WS_TOK_W  = WS_TOK_E + 32768;        // 4096*2 float
static constexpr size_t WS_ROW_T  = WS_TOK_W + 32768;        // 8704 int
static constexpr size_t WS_ROW_W  = WS_ROW_T + 34816;        // 8704 float
static constexpr size_t WS_H      = WS_ROW_W + 34816;        // 8704*5632 bf16 (~98 MB), 16B aligned

typedef __attribute__((ext_vector_type(16))) __bf16 v16bf;
typedef __attribute__((ext_vector_type(8)))  float  v8f;
union Frag { v16bf v; uint4 q[2]; };

// ---------------- K0: zero routing metadata ----------------
__global__ void k0_zero_meta(int* meta) {
    int i = blockIdx.x * blockDim.x + threadIdx.x;
    if (i < (int)(WS_TOK_E / 4)) meta[i] = 0;
}

// ---------------- K1: gating (softmax + top-2) ----------------
__global__ void __launch_bounds__(64)
k1_gate(const float* __restrict__ x, const float* __restrict__ Wg,
        int* __restrict__ tok_e, float* __restrict__ tok_w, int* __restrict__ counts) {
    __shared__ float red[64][NEXP];
    const int t = blockIdx.x;
    const float* xr = x + (size_t)t * D;
    float p[NEXP];
#pragma unroll
    for (int e = 0; e < NEXP; ++e) p[e] = 0.f;
    for (int k = threadIdx.x; k < D; k += 64) {
        float xv = xr[k];
#pragma unroll
        for (int e = 0; e < NEXP; ++e) p[e] += xv * Wg[k * NEXP + e];
    }
#pragma unroll
    for (int e = 0; e < NEXP; ++e) red[threadIdx.x][e] = p[e];
    __syncthreads();
    if (threadIdx.x == 0) {
        float l[NEXP];
#pragma unroll
        for (int e = 0; e < NEXP; ++e) {
            float s = 0.f;
            for (int i = 0; i < 64; ++i) s += red[i][e];
            l[e] = s;
        }
        float mx = l[0];
#pragma unroll
        for (int e = 1; e < NEXP; ++e) mx = fmaxf(mx, l[e]);
        float w[NEXP];
#pragma unroll
        for (int e = 0; e < NEXP; ++e) w[e] = __expf(l[e] - mx);
        int e0 = 0;
#pragma unroll
        for (int e = 1; e < NEXP; ++e) if (w[e] > w[e0]) e0 = e;
        int e1 = (e0 == 0) ? 1 : 0;
#pragma unroll
        for (int e = 0; e < NEXP; ++e) if (e != e0 && w[e] > w[e1]) e1 = e;
        float inv = 1.f / (w[e0] + w[e1]);
        tok_e[t * 2 + 0] = e0; tok_e[t * 2 + 1] = e1;
        tok_w[t * 2 + 0] = w[e0] * inv; tok_w[t * 2 + 1] = w[e1] * inv;
        atomicAdd(&counts[e0], 1);
        atomicAdd(&counts[e1], 1);
    }
}

// ---------------- K2: single-thread scan (padded offsets, tile->expert) ----------------
__global__ void k2_scan(const int* __restrict__ counts, int* __restrict__ cursor,
                        int* __restrict__ tot, int* __restrict__ tile_e,
                        int* __restrict__ row_t, float* __restrict__ row_w) {
    if (threadIdx.x != 0 || blockIdx.x != 0) return;
    int base = 0, nt = 0;
    for (int e = 0; e < NEXP; ++e) {
        int cnt = counts[e];
        cursor[e] = base;
        int tiles = (cnt + BM - 1) / BM;
        for (int tt = 0; tt < tiles; ++tt) tile_e[nt + tt] = e;
        for (int s = cnt; s < tiles * BM; ++s) { row_t[base + s] = 0; row_w[base + s] = 0.f; }
        base += tiles * BM;
        nt += tiles;
    }
    *tot = nt;
}

// ---------------- K3: scatter tokens into grouped rows ----------------
__global__ void k3_scatter(const int* __restrict__ tok_e, const float* __restrict__ tok_w,
                           int* __restrict__ cursor, int* __restrict__ row_t,
                           float* __restrict__ row_w) {
    int t = blockIdx.x * blockDim.x + threadIdx.x;
    if (t >= TOKENS) return;
#pragma unroll
    for (int j = 0; j < 2; ++j) {
        int e = tok_e[t * 2 + j];
        int slot = atomicAdd(&cursor[e], 1);
        row_t[slot] = t;
        row_w[slot] = tok_w[t * 2 + j];
    }
}

// ---------------- K4: zero output ----------------
__global__ void k4_zero(float4* __restrict__ out) {
    out[blockIdx.x * blockDim.x + threadIdx.x] = make_float4(0.f, 0.f, 0.f, 0.f);
}

// ---------------- K5: pass-1 grouped GEMM  H = silu(x*W1) * (x*W3) ----------------
__global__ void __launch_bounds__(NTHREADS)
k5_ffn1(const float* __restrict__ x, const float* __restrict__ W1g,
        const float* __restrict__ W3g, const int* __restrict__ row_t,
        const int* __restrict__ tile_e, const int* __restrict__ tot,
        __bf16* __restrict__ H) {
    const int mt = blockIdx.x;
    if (mt >= *tot) return;
    const int e = tile_e[mt];
    const int rowbase = mt * BM;
    const int n0 = blockIdx.y * BN;
    const float* __restrict__ B1 = W1g + (size_t)e * D * HID;
    const float* __restrict__ B3 = W3g + (size_t)e * D * HID;

    __shared__ __align__(16) __bf16 As[BM][LDA];
    __shared__ __align__(16) __bf16 Bs1[BN][LDB];
    __shared__ __align__(16) __bf16 Bs3[BN][LDB];

    const int tid  = threadIdx.x;
    const int lane = tid & 31;
    const int wid  = tid >> 5;           // 0..7
    const int lh   = lane & 15;
    const int hi   = lane >> 4;          // half-wave select
    const int wm   = (wid >> 1) * 16;    // wave M offset: 0/16/32/48
    const int wn   = (wid & 1) * 64;     // wave N offset: 0/64

    // A loader: 4 threads per row, two float4 each (64 rows x 32 k fp32)
    const int am = tid >> 2, aq = tid & 3;
    const float* __restrict__ aptr = x + (size_t)row_t[rowbase + am] * D;
    // B loader: lane picks 4 consecutive n, wid picks k start (32 k x 128 n fp32)
    const int bn4 = lane * 4, bk0 = wid;

    v8f accg[4], accv[4];
#pragma unroll
    for (int nn = 0; nn < 4; ++nn)
#pragma unroll
        for (int r = 0; r < 8; ++r) { accg[nn][r] = 0.f; accv[nn][r] = 0.f; }

    for (int k0 = 0; k0 < D; k0 += BK) {
        // ---- stage A (fp32 -> bf16, k-contiguous) ----
#pragma unroll
        for (int i = 0; i < 2; ++i) {
            int kk = aq * 4 + i * 16;
            float4 f = *(const float4*)(aptr + k0 + kk);
            As[am][kk + 0] = (__bf16)f.x; As[am][kk + 1] = (__bf16)f.y;
            As[am][kk + 2] = (__bf16)f.z; As[am][kk + 3] = (__bf16)f.w;
        }
        // ---- stage B1/B3 transposed into LDS (n-major, k-contiguous) ----
#pragma unroll
        for (int i = 0; i < 4; ++i) {
            int kk = bk0 + i * 8;
            const float4 f1 = *(const float4*)(B1 + (size_t)(k0 + kk) * HID + n0 + bn4);
            const float4 f3 = *(const float4*)(B3 + (size_t)(k0 + kk) * HID + n0 + bn4);
            Bs1[bn4 + 0][kk] = (__bf16)f1.x; Bs1[bn4 + 1][kk] = (__bf16)f1.y;
            Bs1[bn4 + 2][kk] = (__bf16)f1.z; Bs1[bn4 + 3][kk] = (__bf16)f1.w;
            Bs3[bn4 + 0][kk] = (__bf16)f3.x; Bs3[bn4 + 1][kk] = (__bf16)f3.y;
            Bs3[bn4 + 2][kk] = (__bf16)f3.z; Bs3[bn4 + 3][kk] = (__bf16)f3.w;
        }
        if (k0 + BK < D) {  // prefetch next weight K-tile (global_prefetch_b8)
            __builtin_prefetch(B1 + (size_t)(k0 + BK + bk0) * HID + n0 + bn4, 0, 0);
            __builtin_prefetch(B3 + (size_t)(k0 + BK + bk0) * HID + n0 + bn4, 0, 0);
        }
        __syncthreads();

        // ---- A fragment: 2x ds_load_b128 (rows are 80B, 16B aligned) ----
        Frag af;
        af.q[0] = *(const uint4*)&As[wm + lh][hi * 16];
        af.q[1] = *(const uint4*)&As[wm + lh][hi * 16 + 8];
#pragma unroll
        for (int nn = 0; nn < 4; ++nn) {
            Frag bf1, bf3;
            const int col = wn + nn * 16 + lh;
            bf1.q[0] = *(const uint4*)&Bs1[col][hi * 16];
            bf1.q[1] = *(const uint4*)&Bs1[col][hi * 16 + 8];
            bf3.q[0] = *(const uint4*)&Bs3[col][hi * 16];
            bf3.q[1] = *(const uint4*)&Bs3[col][hi * 16 + 8];
            accg[nn] = __builtin_amdgcn_wmma_f32_16x16x32_bf16(
                false, af.v, false, bf1.v, (short)0, accg[nn], false, false);
            accv[nn] = __builtin_amdgcn_wmma_f32_16x16x32_bf16(
                false, af.v, false, bf3.v, (short)0, accv[nn], false, false);
        }
        __syncthreads();
    }

    // ---- epilogue: silu(g)*v -> bf16 H (grouped rows) ----
#pragma unroll
    for (int nn = 0; nn < 4; ++nn) {
        const int N = n0 + wn + nn * 16 + lh;
#pragma unroll
        for (int r = 0; r < 8; ++r) {
            const int M = wm + r + hi * 8;
            float g = accg[nn][r];
            float s = g / (1.0f + __expf(-g));
            H[(size_t)(rowbase + M) * HID + N] = (__bf16)(s * accv[nn][r]);
        }
    }
}

// ---------------- K6: pass-2 grouped GEMM  out += w_row * (H * W2) ----------------
__global__ void __launch_bounds__(NTHREADS)
k6_ffn2(const __bf16* __restrict__ H, const float* __restrict__ W2g,
        const int* __restrict__ row_t, const float* __restrict__ row_w,
        const int* __restrict__ tile_e, const int* __restrict__ tot,
        float* __restrict__ out) {
    const int mt = blockIdx.x;
    if (mt >= *tot) return;
    const int e = tile_e[mt];
    const int rowbase = mt * BM;
    const int n0 = blockIdx.y * BN;
    const float* __restrict__ B = W2g + (size_t)e * HID * D;

    __shared__ __align__(16) __bf16 As[BM][LDA];
    __shared__ __align__(16) __bf16 Bs[BN][LDB];

    const int tid  = threadIdx.x;
    const int lane = tid & 31;
    const int wid  = tid >> 5;
    const int lh   = lane & 15;
    const int hi   = lane >> 4;
    const int wm   = (wid >> 1) * 16;
    const int wn   = (wid & 1) * 64;

    const int am = tid >> 2, aq = tid & 3;               // A loader (async, 16B/thread)
    const __bf16* __restrict__ aptr = H + (size_t)(rowbase + am) * HID;
    // LDS byte offset of this thread's A-tile destination (flat addr low 32 bits = LDS offset)
    const unsigned a_lds = (unsigned)(uintptr_t)&As[am][aq * 8];
    const int bn4 = lane * 4, bk0 = wid;                 // B loader

    v8f acc[4];
#pragma unroll
    for (int nn = 0; nn < 4; ++nn)
#pragma unroll
        for (int r = 0; r < 8; ++r) acc[nn][r] = 0.f;

    for (int k0 = 0; k0 < HID; k0 += BK) {
        // ---- A tile (already bf16): async global->LDS, no VGPR round-trip ----
        {
            const void* gsrc = (const void*)(aptr + k0 + aq * 8);
            asm volatile("global_load_async_to_lds_b128 %0, %1, off"
                         :: "v"(a_lds), "v"(gsrc) : "memory");
        }
        // ---- B tile (fp32 -> bf16) staged through VGPRs, transposed ----
#pragma unroll
        for (int i = 0; i < 4; ++i) {
            int kk = bk0 + i * 8;
            const float4 f = *(const float4*)(B + (size_t)(k0 + kk) * D + n0 + bn4);
            Bs[bn4 + 0][kk] = (__bf16)f.x; Bs[bn4 + 1][kk] = (__bf16)f.y;
            Bs[bn4 + 2][kk] = (__bf16)f.z; Bs[bn4 + 3][kk] = (__bf16)f.w;
        }
        if (k0 + BK < HID)
            __builtin_prefetch(B + (size_t)(k0 + BK + bk0) * D + n0 + bn4, 0, 0);
        asm volatile("s_wait_asynccnt 0" ::: "memory");
        __syncthreads();

        Frag af;
        af.q[0] = *(const uint4*)&As[wm + lh][hi * 16];
        af.q[1] = *(const uint4*)&As[wm + lh][hi * 16 + 8];
#pragma unroll
        for (int nn = 0; nn < 4; ++nn) {
            Frag bf;
            const int col = wn + nn * 16 + lh;
            bf.q[0] = *(const uint4*)&Bs[col][hi * 16];
            bf.q[1] = *(const uint4*)&Bs[col][hi * 16 + 8];
            acc[nn] = __builtin_amdgcn_wmma_f32_16x16x32_bf16(
                false, af.v, false, bf.v, (short)0, acc[nn], false, false);
        }
        __syncthreads();
    }

    // per-row token id + combine weight (padding rows have w==0 -> add exact 0)
    int   tk[8]; float wt[8];
#pragma unroll
    for (int r = 0; r < 8; ++r) {
        int rowg = rowbase + wm + r + hi * 8;
        tk[r] = row_t[rowg];
        wt[r] = row_w[rowg];
    }
#pragma unroll
    for (int nn = 0; nn < 4; ++nn) {
        const int N = n0 + wn + nn * 16 + lh;
#pragma unroll
        for (int r = 0; r < 8; ++r)
            atomicAdd(&out[(size_t)tk[r] * D + N], wt[r] * acc[nn][r]);
    }
}

// ---------------- host launcher ----------------
extern "C" void kernel_launch(void* const* d_in, const int* in_sizes, int n_in,
                              void* d_out, int out_size, void* d_ws, size_t ws_size,
                              hipStream_t stream) {
    const float* x  = (const float*)d_in[0];
    const float* Wg = (const float*)d_in[1];
    const float* W1 = (const float*)d_in[2];
    const float* W3 = (const float*)d_in[3];
    const float* W2 = (const float*)d_in[4];
    float* out = (float*)d_out;
    char* ws = (char*)d_ws;

    int*    counts = (int*)(ws + WS_COUNTS);
    int*    cursor = (int*)(ws + WS_CURSOR);
    int*    tot    = (int*)(ws + WS_TOT);
    int*    tile_e = (int*)(ws + WS_TILE_E);
    int*    tok_e  = (int*)(ws + WS_TOK_E);
    float*  tok_w  = (float*)(ws + WS_TOK_W);
    int*    row_t  = (int*)(ws + WS_ROW_T);
    float*  row_w  = (float*)(ws + WS_ROW_W);
    __bf16* H      = (__bf16*)(ws + WS_H);

    k0_zero_meta<<<2, 256, 0, stream>>>((int*)ws);
    k1_gate<<<TOKENS, 64, 0, stream>>>(x, Wg, tok_e, tok_w, counts);
    k2_scan<<<1, 1, 0, stream>>>(counts, cursor, tot, tile_e, row_t, row_w);
    k3_scatter<<<TOKENS / 256, 256, 0, stream>>>(tok_e, tok_w, cursor, row_t, row_w);
    k4_zero<<<(TOKENS * D / 4) / 256, 256, 0, stream>>>((float4*)out);
    k5_ffn1<<<dim3(MAX_TILES, HID / BN), NTHREADS, 0, stream>>>(x, W1, W3, row_t, tile_e, tot, H);
    k6_ffn2<<<dim3(MAX_TILES, D / BN), NTHREADS, 0, stream>>>(H, W2, row_t, row_w, tile_e, tot, out);
}
